// BigramNeuralNetwork_20109036879947
// MI455X (gfx1250) — matrix operand order repaired and checked by reading, not compiled, not run on hardware
//
#include <hip/hip_runtime.h>
#include <hip/hip_bf16.h>

typedef _Float16 v16h __attribute__((ext_vector_type(16)));
typedef _Float16 v8h  __attribute__((ext_vector_type(8)));
typedef float    v8f  __attribute__((ext_vector_type(8)));

#define NB   16
#define TT   256
#define BT   4096     // B*T
#define DM   384
#define DI   768
#define DST  16
#define DTR  24
#define DFF  1536
#define NV   32000
#define NL   6

#define GF_BIAS  1
#define GF_RELU  2
#define GF_ACC   4
#define GF_OUTH  8

// async global -> LDS copy (CDNA5, ASYNCcnt-tracked). ldsoff: LDS byte offset VGPR,
// gptr: 64-bit global address.
#define ASYNC_LDS_B128(ldsoff, gptr)                                             \
    asm volatile("global_load_async_to_lds_b128 %0, %1, off"                     \
                 :: "v"(ldsoff), "v"((unsigned long long)(uintptr_t)(gptr))      \
                 : "memory")
#define WAIT_ASYNC0() asm volatile("s_wait_asynccnt 0x0" ::: "memory")

// ---------------- generic f32 -> f16 convert ----------------
__global__ void f32_to_f16_kernel(const float* __restrict__ in, _Float16* __restrict__ out, size_t n) {
    size_t i = (size_t)blockIdx.x * blockDim.x + threadIdx.x;
    size_t stride = (size_t)gridDim.x * blockDim.x;
    for (; i < n; i += stride) out[i] = (_Float16)in[i];
}

// ---------------- embedding ----------------
__global__ void embed_kernel(const int* __restrict__ idx, const float* __restrict__ tok,
                             const float* __restrict__ pos, float* __restrict__ x) {
    int i = blockIdx.x * blockDim.x + threadIdx.x;
    if (i >= BT * DM) return;
    int row = i / DM, col = i - row * DM;
    int t = row & (TT - 1);
    x[i] = tok[(size_t)idx[row] * DM + col] + pos[t * DM + col];
}

// ---------------- LayerNorm (wave-per-row, wave32 shuffles), f16 output ----------------
__global__ __launch_bounds__(256)
void layernorm_f16_kernel(const float* __restrict__ x, const float* __restrict__ g,
                          const float* __restrict__ b, _Float16* __restrict__ out) {
    int wave = threadIdx.x >> 5, lane = threadIdx.x & 31;
    int row = blockIdx.x * 8 + wave;
    if (row >= BT) return;
    const float* xr = x + (size_t)row * DM;
    float vals[DM / 32];
    float s = 0.f;
#pragma unroll
    for (int i = 0; i < DM / 32; ++i) { vals[i] = xr[lane + 32 * i]; s += vals[i]; }
#pragma unroll
    for (int off = 16; off; off >>= 1) s += __shfl_xor(s, off, 32);
    float mean = s * (1.0f / DM);
    float v = 0.f;
#pragma unroll
    for (int i = 0; i < DM / 32; ++i) { float d = vals[i] - mean; v += d * d; }
#pragma unroll
    for (int off = 16; off; off >>= 1) v += __shfl_xor(v, off, 32);
    float rstd = rsqrtf(v * (1.0f / DM) + 1e-5f);
    _Float16* orow = out + (size_t)row * DM;
#pragma unroll
    for (int i = 0; i < DM / 32; ++i) {
        int c = lane + 32 * i;
        orow[c] = (_Float16)((vals[i] - mean) * rstd * g[c] + b[c]);
    }
}

// ---------------- WMMA f16 GEMM with async-LDS double buffering ----------------
// out[M,N] = A[M,K] * W[N,K]^T (+bias)(relu)(+=)(f16 out)
// Block tile 64(M) x 128(N) x 32(K). 8 waves, each owns a 32x32 C sub-tile (4 WMMAs/step).
// Tiles are copied global->LDS with GLOBAL_LOAD_ASYNC_TO_LDS_B128 (ASYNCcnt), double-buffered
// so the copy of tile k+1 overlaps WMMA on tile k. Fragments come from LDS (ds_load_b128).
// grid = (N/128, M/64), block = 256. Requires M%64==0, N%128==0, K%32==0.
__global__ __launch_bounds__(256)
void gemm_wmma_f16(const _Float16* __restrict__ A, const _Float16* __restrict__ W,
                   const float* __restrict__ bias, float* __restrict__ outf,
                   _Float16* __restrict__ outh, int M, int N, int K, int flags) {
    __shared__ __align__(16) _Float16 sA[2][64 * 32];
    __shared__ __align__(16) _Float16 sB[2][128 * 32];

    const int tid  = threadIdx.x;
    const int lane = tid & 31;
    const int wave = tid >> 5;
    const int mW = (wave & 1) * 32;     // wave M offset in block tile
    const int nW = (wave >> 1) * 32;    // wave N offset in block tile
    const int mTile = blockIdx.y * 64;
    const int nTile = blockIdx.x * 128;
    const int ml = lane & 15;           // row/col within 16x16 tile
    const int kh = lane >> 4;           // K-half selector

    // cooperative copy mapping: chunks of 8 halves (16B)
    const int arow = tid >> 2, ac = (tid & 3) * 8;          // A: 256 chunks, 1/thread
    const int brow1 = (tid + 256) >> 2;                     // B: 512 chunks, 2/thread
    const unsigned aoff  = (unsigned)(arow * 32 + ac) * 2u;
    const unsigned boff0 = aoff;                            // same row/chunk mapping
    const unsigned boff1 = (unsigned)(brow1 * 32 + ac) * 2u;
    const unsigned sAb[2] = { (unsigned)(uintptr_t)&sA[0][0], (unsigned)(uintptr_t)&sA[1][0] };
    const unsigned sBb[2] = { (unsigned)(uintptr_t)&sB[0][0], (unsigned)(uintptr_t)&sB[1][0] };

    const _Float16* Ag  = A + (size_t)(mTile + arow) * K + ac;
    const _Float16* Bg0 = W + (size_t)(nTile + arow) * K + ac;
    const _Float16* Bg1 = W + (size_t)(nTile + brow1) * K + ac;

    const int nK = K >> 5;

    // issue async copies for K-step kt into buffer buf
    auto issue = [&](int kt, int buf) {
        const int k0 = kt * 32;
        ASYNC_LDS_B128(sAb[buf] + aoff,  Ag  + k0);
        ASYNC_LDS_B128(sBb[buf] + boff0, Bg0 + k0);
        ASYNC_LDS_B128(sBb[buf] + boff1, Bg1 + k0);
    };

    v8f acc[2][2];
#pragma unroll
    for (int i = 0; i < 2; ++i)
#pragma unroll
        for (int j = 0; j < 2; ++j) acc[i][j] = v8f{};

    issue(0, 0);
    WAIT_ASYNC0();
    __syncthreads();

    for (int kt = 0; kt < nK; ++kt) {
        const int cur = kt & 1;
        if (kt + 1 < nK) issue(kt + 1, cur ^ 1);   // overlap copy with WMMA below

        const _Float16* a0 = &sA[cur][0];
        const _Float16* b0 = &sB[cur][0];

        v16h af[2], bf[2];
#pragma unroll
        for (int i = 0; i < 2; ++i) {
            // A fragment (ISA 16-bit A 16x32): VGPR0-3 -> K in [kh*8,kh*8+8), VGPR4-7 -> +16
            const _Float16* ap = a0 + (mW + i * 16 + ml) * 32 + kh * 8;
            *(v8h*)&af[i]       = *(const v8h*)(ap);
            *((v8h*)&af[i] + 1) = *(const v8h*)(ap + 16);
            // B fragment: same K order as A: VGPR0-3 -> K in [kh*8,kh*8+8), VGPR4-7 -> +16
            const _Float16* bp = b0 + (nW + i * 16 + ml) * 32 + kh * 8;
            *(v8h*)&bf[i]       = *(const v8h*)(bp);
            *((v8h*)&bf[i] + 1) = *(const v8h*)(bp + 16);
        }
#pragma unroll
        for (int i = 0; i < 2; ++i)
#pragma unroll
            for (int j = 0; j < 2; ++j)
                acc[i][j] = __builtin_amdgcn_wmma_f32_16x16x32_f16(
                    false, af[i], false, bf[j], (short)0, acc[i][j], false, false);

        WAIT_ASYNC0();      // next-tile copies (issued above) are now complete
        __syncthreads();
    }

    // epilogue: lane owns column n, rows m = 8*kh + r (ISA C/D layout)
#pragma unroll
    for (int j = 0; j < 2; ++j) {
        const int n = nTile + nW + j * 16 + ml;
        const float bv = (flags & GF_BIAS) ? bias[n] : 0.0f;
#pragma unroll
        for (int i = 0; i < 2; ++i) {
#pragma unroll
            for (int r = 0; r < 8; ++r) {
                int m = mTile + mW + i * 16 + kh * 8 + r;
                float v = acc[i][j][r] + bv;
                if (flags & GF_RELU) v = fmaxf(v, 0.0f);
                size_t o = (size_t)m * N + n;
                if (flags & GF_OUTH)      outh[o] = (_Float16)v;
                else if (flags & GF_ACC)  outf[o] += v;
                else                      outf[o] = v;
            }
        }
    }
}

// ---------------- causal depthwise conv1d + bias + SiLU ----------------
__global__ void conv_silu_kernel(const float* __restrict__ xz, const float* __restrict__ cw,
                                 const float* __restrict__ cb, float* __restrict__ xc) {
    int i = blockIdx.x * blockDim.x + threadIdx.x;
    if (i >= BT * DI) return;
    int row = i / DI, d = i - row * DI;
    int b = row / TT, t = row - b * TT;
    float acc = cb[d];
#pragma unroll
    for (int j = 0; j < 4; ++j) {
        int tt = t - 3 + j;
        if (tt >= 0) acc += cw[d * 4 + j] * xz[(size_t)(b * TT + tt) * (2 * DI) + d];
    }
    xc[i] = acc / (1.0f + __expf(-acc));
}

// ---------------- x_proj: [BT,768] x [56,768]^T (small N -> VALU) ----------------
__global__ void xproj_kernel(const float* __restrict__ xc, const float* __restrict__ w,
                             float* __restrict__ xdbl) {
    int i = blockIdx.x * blockDim.x + threadIdx.x;
    if (i >= BT * (DTR + 2 * DST)) return;
    int row = i / (DTR + 2 * DST), n = i - row * (DTR + 2 * DST);
    const float* a  = xc + (size_t)row * DI;
    const float* wr = w + (size_t)n * DI;
    float acc = 0.f;
    for (int k = 0; k < DI; ++k) acc += a[k] * wr[k];
    xdbl[i] = acc;
}

// ---------------- dt_proj + softplus (K=24 -> VALU) ----------------
__global__ void dtproj_kernel(const float* __restrict__ xdbl, const float* __restrict__ w,
                              const float* __restrict__ bias, float* __restrict__ dt) {
    int i = blockIdx.x * blockDim.x + threadIdx.x;
    if (i >= BT * DI) return;
    int row = i / DI, d = i - row * DI;
    const float* a  = xdbl + (size_t)row * (DTR + 2 * DST);
    const float* wr = w + (size_t)d * DTR;
    float acc = bias[d];
#pragma unroll
    for (int k = 0; k < DTR; ++k) acc += a[k] * wr[k];
    dt[i] = (acc > 20.f) ? acc : log1pf(__expf(acc));
}

// ---------------- selective scan, one thread per (b, d, state) ----------------
// h_t = h*exp(dt*A) + dt*B_t*x_t ; y = sum_s h*C_t  (reduce over 16 state lanes)
// block = 256 threads = 16 d-channels x 16 states; grid = NB * (DI/16)
__global__ __launch_bounds__(256)
void scan_kernel(const float* __restrict__ xdbl, const float* __restrict__ dt,
                 const float* __restrict__ xc, const float* __restrict__ A_log,
                 float* __restrict__ y) {
    const int b  = blockIdx.x / (DI / 16);
    const int dg = blockIdx.x % (DI / 16);
    const int s  = threadIdx.x & 15;
    const int dl = threadIdx.x >> 4;
    const int d  = dg * 16 + dl;

    const float Av = -__expf(A_log[(size_t)d * DST + s]);
    float h = 0.f;
    for (int t = 0; t < TT; ++t) {
        const size_t row = (size_t)b * TT + t;
        const float dtv = dt[row * DI + d];           // broadcast across 16 s-lanes
        const float xv  = xc[row * DI + d];
        const float Bv  = xdbl[row * (DTR + 2 * DST) + DTR + s];
        const float Cv  = xdbl[row * (DTR + 2 * DST) + DTR + DST + s];
        h = h * __expf(dtv * Av) + dtv * Bv * xv;
        float yv = h * Cv;
#pragma unroll
        for (int off = 8; off; off >>= 1) yv += __shfl_xor(yv, off, 16);
        if (s == 0) y[row * DI + d] = yv;
    }
}

// ---------------- D-skip + SiLU gate, f16 out for out_proj GEMM ----------------
__global__ void gate_kernel(const float* __restrict__ y, const float* __restrict__ xc,
                            const float* __restrict__ xz, const float* __restrict__ Dp,
                            _Float16* __restrict__ y16) {
    int i = blockIdx.x * blockDim.x + threadIdx.x;
    if (i >= BT * DI) return;
    int row = i / DI, d = i - row * DI;
    float z = xz[(size_t)row * (2 * DI) + DI + d];
    float sz = z / (1.0f + __expf(-z));
    y16[i] = (_Float16)((y[i] + xc[i] * Dp[d]) * sz);
}

// ---------------- per-row NLL (stable log-softmax) ----------------
__global__ __launch_bounds__(256)
void nll_kernel(const float* __restrict__ logits, const int* __restrict__ targets,
                float* __restrict__ nll) {
    int row = blockIdx.x;
    const float* lr = logits + (size_t)row * NV;
    __shared__ float red[256];
    float mx = -1e30f;
    for (int c = threadIdx.x; c < NV; c += 256) mx = fmaxf(mx, lr[c]);
    red[threadIdx.x] = mx; __syncthreads();
    for (int s = 128; s; s >>= 1) {
        if (threadIdx.x < s) red[threadIdx.x] = fmaxf(red[threadIdx.x], red[threadIdx.x + s]);
        __syncthreads();
    }
    mx = red[0]; __syncthreads();
    float sum = 0.f;
    for (int c = threadIdx.x; c < NV; c += 256) sum += __expf(lr[c] - mx);
    red[threadIdx.x] = sum; __syncthreads();
    for (int s = 128; s; s >>= 1) {
        if (threadIdx.x < s) red[threadIdx.x] += red[threadIdx.x + s];
        __syncthreads();
    }
    if (threadIdx.x == 0) nll[row] = logf(red[0]) + mx - lr[targets[row]];
}

__global__ __launch_bounds__(256)
void loss_kernel(const float* __restrict__ nll, float* __restrict__ out) {
    __shared__ float red[256];
    float s = 0.f;
    for (int i = threadIdx.x; i < BT; i += 256) s += nll[i];
    red[threadIdx.x] = s; __syncthreads();
    for (int k = 128; k; k >>= 1) {
        if (threadIdx.x < k) red[threadIdx.x] += red[threadIdx.x + k];
        __syncthreads();
    }
    if (threadIdx.x == 0) out[0] = red[0] * (1.0f / BT);
}

// ==================================================================
static inline char* wsal(char*& p, size_t bytes) {
    char* r = p;
    p += (bytes + 255) & ~(size_t)255;
    return r;
}

extern "C" void kernel_launch(void* const* d_in, const int* in_sizes, int n_in,
                              void* d_out, int out_size, void* d_ws, size_t ws_size,
                              hipStream_t stream) {
    const float* tok_emb  = (const float*)d_in[0];
    const float* pos_emb  = (const float*)d_in[1];
    const float* ln1_g    = (const float*)d_in[2];
    const float* ln1_b    = (const float*)d_in[3];
    const float* in_w     = (const float*)d_in[4];
    const float* conv_w   = (const float*)d_in[5];
    const float* conv_b   = (const float*)d_in[6];
    const float* xp_w     = (const float*)d_in[7];
    const float* dtp_w    = (const float*)d_in[8];
    const float* dtp_b    = (const float*)d_in[9];
    const float* A_log    = (const float*)d_in[10];
    const float* Dp       = (const float*)d_in[11];
    const float* out_w    = (const float*)d_in[12];
    const float* ln2_g    = (const float*)d_in[13];
    const float* ln2_b    = (const float*)d_in[14];
    const float* ffn_w1   = (const float*)d_in[15];
    const float* ffn_b1   = (const float*)d_in[16];
    const float* ffn_w2   = (const float*)d_in[17];
    const float* ffn_b2   = (const float*)d_in[18];
    const float* lm_w     = (const float*)d_in[19];
    const float* lm_b     = (const float*)d_in[20];
    const int*   idx      = (const int*)d_in[21];
    const int*   targets  = (const int*)d_in[22];

    float* logits = (float*)d_out;                 // [BT, V]
    float* loss   = logits + (size_t)BT * NV;      // final element

    char* p = (char*)d_ws;
    float*    s_x     = (float*)   wsal(p, (size_t)BT * DM * 4);
    _Float16* s_ln16  = (_Float16*)wsal(p, (size_t)BT * DM * 2);
    float*    s_xz    = (float*)   wsal(p, (size_t)BT * 2 * DI * 4);
    float*    s_xc    = (float*)   wsal(p, (size_t)BT * DI * 4);
    float*    s_xdbl  = (float*)   wsal(p, (size_t)BT * (DTR + 2 * DST) * 4);
    float*    s_dt    = (float*)   wsal(p, (size_t)BT * DI * 4);
    float*    s_y     = (float*)   wsal(p, (size_t)BT * DI * 4);
    _Float16* s_y16   = (_Float16*)wsal(p, (size_t)BT * DI * 2);
    _Float16* s_ffh16 = (_Float16*)wsal(p, (size_t)BT * DFF * 2);
    _Float16* s_x16   = (_Float16*)wsal(p, (size_t)BT * DM * 2);
    float*    s_nll   = (float*)   wsal(p, (size_t)BT * 4);
    _Float16* w16_in  = (_Float16*)wsal(p, (size_t)NL * 2 * DI * DM * 2);
    _Float16* w16_out = (_Float16*)wsal(p, (size_t)NL * DM * DI * 2);
    _Float16* w16_f1  = (_Float16*)wsal(p, (size_t)NL * DFF * DM * 2);
    _Float16* w16_f2  = (_Float16*)wsal(p, (size_t)NL * DM * DFF * 2);
    _Float16* w16_lm  = (_Float16*)wsal(p, (size_t)NV * DM * 2);

    // --- weight conversion to f16 (deterministic, every call) ---
    {
        size_t n;
        n = (size_t)NL * 2 * DI * DM;
        hipLaunchKernelGGL(f32_to_f16_kernel, dim3(2048), dim3(256), 0, stream, in_w, w16_in, n);
        n = (size_t)NL * DM * DI;
        hipLaunchKernelGGL(f32_to_f16_kernel, dim3(2048), dim3(256), 0, stream, out_w, w16_out, n);
        n = (size_t)NL * DFF * DM;
        hipLaunchKernelGGL(f32_to_f16_kernel, dim3(2048), dim3(256), 0, stream, ffn_w1, w16_f1, n);
        n = (size_t)NL * DM * DFF;
        hipLaunchKernelGGL(f32_to_f16_kernel, dim3(2048), dim3(256), 0, stream, ffn_w2, w16_f2, n);
        n = (size_t)NV * DM;
        hipLaunchKernelGGL(f32_to_f16_kernel, dim3(4096), dim3(256), 0, stream, lm_w, w16_lm, n);
    }

    // --- embedding ---
    hipLaunchKernelGGL(embed_kernel, dim3((BT * DM + 255) / 256), dim3(256), 0, stream,
                       idx, tok_emb, pos_emb, s_x);

    const dim3 blk(256);
    const dim3 lnGrid(BT / 8);
    const dim3 ewDI((BT * DI + 255) / 256);

    for (int l = 0; l < NL; ++l) {
        // LN1 -> f16
        hipLaunchKernelGGL(layernorm_f16_kernel, lnGrid, blk, 0, stream,
                           s_x, ln1_g + l * DM, ln1_b + l * DM, s_ln16);
        // in_proj: [BT,384] -> [BT,1536]
        hipLaunchKernelGGL(gemm_wmma_f16, dim3(2 * DI / 128, BT / 64), blk, 0, stream,
                           s_ln16, w16_in + (size_t)l * 2 * DI * DM, (const float*)nullptr,
                           s_xz, (_Float16*)nullptr, BT, 2 * DI, DM, 0);
        // causal depthwise conv + SiLU
        hipLaunchKernelGGL(conv_silu_kernel, ewDI, blk, 0, stream,
                           s_xz, conv_w + (size_t)l * DI * 4, conv_b + (size_t)l * DI, s_xc);
        // x_proj (small N=56)
        hipLaunchKernelGGL(xproj_kernel, dim3((BT * (DTR + 2 * DST) + 255) / 256), blk, 0, stream,
                           s_xc, xp_w + (size_t)l * (DTR + 2 * DST) * DI, s_xdbl);
        // dt_proj + softplus (small K=24)
        hipLaunchKernelGGL(dtproj_kernel, ewDI, blk, 0, stream,
                           s_xdbl, dtp_w + (size_t)l * DI * DTR, dtp_b + (size_t)l * DI, s_dt);
        // selective scan
        hipLaunchKernelGGL(scan_kernel, dim3(NB * (DI / 16)), blk, 0, stream,
                           s_xdbl, s_dt, s_xc, A_log + (size_t)l * DI * DST, s_y);
        // D skip + gate -> f16
        hipLaunchKernelGGL(gate_kernel, ewDI, blk, 0, stream,
                           s_y, s_xc, s_xz, Dp + (size_t)l * DI, s_y16);
        // out_proj accumulate into residual: x += y @ out_w^T
        hipLaunchKernelGGL(gemm_wmma_f16, dim3(DM / 128, BT / 64), blk, 0, stream,
                           s_y16, w16_out + (size_t)l * DM * DI, (const float*)nullptr,
                           s_x, (_Float16*)nullptr, BT, DM, DI, GF_ACC);
        // LN2 -> f16
        hipLaunchKernelGGL(layernorm_f16_kernel, lnGrid, blk, 0, stream,
                           s_x, ln2_g + l * DM, ln2_b + l * DM, s_ln16);
        // FFN1 + bias + relu -> f16
        hipLaunchKernelGGL(gemm_wmma_f16, dim3(DFF / 128, BT / 64), blk, 0, stream,
                           s_ln16, w16_f1 + (size_t)l * DFF * DM, ffn_b1 + (size_t)l * DFF,
                           (float*)nullptr, s_ffh16, BT, DFF, DM, GF_BIAS | GF_RELU | GF_OUTH);
        // FFN2 + bias, accumulate into residual
        hipLaunchKernelGGL(gemm_wmma_f16, dim3(DM / 128, BT / 64), blk, 0, stream,
                           s_ffh16, w16_f2 + (size_t)l * DM * DFF, ffn_b2 + (size_t)l * DM,
                           s_x, (_Float16*)nullptr, BT, DM, DFF, GF_BIAS | GF_ACC);
    }

    // --- lm_head: logits = x @ lm_w^T + lm_b ---
    hipLaunchKernelGGL(f32_to_f16_kernel, dim3(2048), blk, 0, stream,
                       s_x, s_x16, (size_t)BT * DM);
    hipLaunchKernelGGL(gemm_wmma_f16, dim3(NV / 128, BT / 64), blk, 0, stream,
                       s_x16, w16_lm, lm_b, logits, (_Float16*)nullptr, BT, NV, DM, GF_BIAS);

    // --- loss ---
    hipLaunchKernelGGL(nll_kernel, dim3(BT), blk, 0, stream, logits, targets, s_nll);
    hipLaunchKernelGGL(loss_kernel, dim3(1), blk, 0, stream, s_nll, loss);
}
